// IndRNN_56573309223922
// MI455X (gfx1250) — matrix operand order repaired and checked by reading, not compiled, not run on hardware
//
#include <hip/hip_runtime.h>
#include <cstdint>
#include <cstddef>

// ---------------------------------------------------------------------------
// Problem constants (match reference): T=1024, B=64, D=512, H=1024
// ---------------------------------------------------------------------------
static constexpr int  kT = 1024;
static constexpr int  kB = 64;
static constexpr int  kD = 512;
static constexpr int  kH = 1024;
static constexpr long long kM  = (long long)kT * kB;   // 65536 rows for GEMMs
static constexpr long long kBH = (long long)kB * kH;   // 65536 recurrence lanes

typedef __attribute__((ext_vector_type(16))) __bf16 v16bf;
typedef __attribute__((ext_vector_type(8)))  float  v8f;

union Frag16 {
    uint4  u[2];
    v16bf  v;
};

// ---------------------------------------------------------------------------
// bf16 helpers (raw ushort storage; RNE rounding)
// ---------------------------------------------------------------------------
__device__ __forceinline__ unsigned short f32_to_bf16_rne(float f) {
    unsigned int u = __float_as_uint(f);
    u += 0x7FFFu + ((u >> 16) & 1u);
    return (unsigned short)(u >> 16);
}
__device__ __forceinline__ float bf16_to_f32(unsigned short h) {
    return __uint_as_float(((unsigned int)h) << 16);
}

// ---------------------------------------------------------------------------
// f32 -> bf16 (hi) and optional residual (lo) split conversion
// ---------------------------------------------------------------------------
__global__ void to_bf16_split(const float* __restrict__ in,
                              unsigned short* __restrict__ hi,
                              unsigned short* __restrict__ lo,
                              int n, int split) {
    int stride = gridDim.x * blockDim.x;
    for (int i = blockIdx.x * blockDim.x + threadIdx.x; i < n; i += stride) {
        float f = in[i];
        unsigned short h = f32_to_bf16_rne(f);
        hi[i] = h;
        if (split) {
            lo[i] = f32_to_bf16_rne(f - bf16_to_f32(h));
        }
    }
}

// ---------------------------------------------------------------------------
// Fragment loader: two 16-byte global loads per lane, addressed so the data
// lands in one contiguous k order (k = k0 + 16*half + s) for both A and B.
//   A (16x32, MxK):  lane half 0 -> K {k0..k0+15}
//                    lane half 1 -> K {k0+16..k0+31}
//                    chunks at (ka, ka+8),  ka = k0 + half*16
//   B (32x16, KxN):  lane half 0 -> K {k0..k0+15}, half 1 -> {k0+16..k0+31}
//                    chunks at (kb, kb+8),  kb = k0 + half*16
// ---------------------------------------------------------------------------
__device__ __forceinline__ v16bf load_frag(const unsigned short* __restrict__ p,
                                           int ld, int row, int kfirst, int ksecond) {
    Frag16 f;
    const unsigned short* base = p + (size_t)row * (size_t)ld;
    f.u[0] = *reinterpret_cast<const uint4*>(base + kfirst);
    f.u[1] = *reinterpret_cast<const uint4*>(base + ksecond);
    return f.v;
}

// ---------------------------------------------------------------------------
// C[m,n] = sum_k A[m,k] * W[n,k] + bias[n]
// A: [M,K] row-major bf16 (hi/lo), W: [N,K] row-major bf16 (hi/lo), C: [M,N] f32
// Block: 256 threads = 8 waves laid out 2(M) x 4(N); wave tile 32x32;
// workgroup tile 64(M) x 128(N). Grid: (N/128, M/64).
// SPLIT: D ~= Ahi*Bhi + Ahi*Blo + Alo*Bhi  (fp32-class accuracy)
// ---------------------------------------------------------------------------
template <bool SPLIT>
__global__ __launch_bounds__(256)
void gemm_wmma_bf16(const unsigned short* __restrict__ Ah,
                    const unsigned short* __restrict__ Al,
                    const unsigned short* __restrict__ Bh,
                    const unsigned short* __restrict__ Bl,
                    const float* __restrict__ bias,
                    float* __restrict__ C,
                    int M, int N, int K) {
    const int wave = threadIdx.x >> 5;
    const int lane = threadIdx.x & 31;
    const int wm = wave & 1;        // 0..1
    const int wn = wave >> 1;       // 0..3
    const int m0 = blockIdx.y * 64  + wm * 32;
    const int n0 = blockIdx.x * 128 + wn * 32;
    const int l16  = lane & 15;
    const int half = lane >> 4;

    const v8f zero = {0.f, 0.f, 0.f, 0.f, 0.f, 0.f, 0.f, 0.f};
    v8f acc[2][2] = {{zero, zero}, {zero, zero}};

    const int arow0 = m0 + l16;
    const int arow1 = m0 + 16 + l16;
    const int brow0 = n0 + l16;
    const int brow1 = n0 + 16 + l16;

    for (int k0 = 0; k0 < K; k0 += 32) {
        const int ka = k0 + half * 16;   // A chunks at ka and ka+8
        const int kb = k0 + half * 16;   // B chunks at kb and kb+8

        // Issue ALL fragment loads up front so they clause together and the
        // scheduler can hide them under the previous iteration's WMMAs.
        v16bf ah[2], bh[2], al[2], bl[2];
        ah[0] = load_frag(Ah, K, arow0, ka, ka + 8);
        ah[1] = load_frag(Ah, K, arow1, ka, ka + 8);
        bh[0] = load_frag(Bh, K, brow0, kb, kb + 8);
        bh[1] = load_frag(Bh, K, brow1, kb, kb + 8);
        if (SPLIT) {
            al[0] = load_frag(Al, K, arow0, ka, ka + 8);
            al[1] = load_frag(Al, K, arow1, ka, ka + 8);
            bl[0] = load_frag(Bl, K, brow0, kb, kb + 8);
            bl[1] = load_frag(Bl, K, brow1, kb, kb + 8);
        }

#pragma unroll
        for (int i = 0; i < 2; ++i)
#pragma unroll
            for (int j = 0; j < 2; ++j) {
                acc[i][j] = __builtin_amdgcn_wmma_f32_16x16x32_bf16(
                    false, ah[i], false, bh[j], (short)0, acc[i][j], false, false);
                if (SPLIT) {
                    acc[i][j] = __builtin_amdgcn_wmma_f32_16x16x32_bf16(
                        false, ah[i], false, bl[j], (short)0, acc[i][j], false, false);
                    acc[i][j] = __builtin_amdgcn_wmma_f32_16x16x32_bf16(
                        false, al[i], false, bh[j], (short)0, acc[i][j], false, false);
                }
            }
    }

    // Epilogue: C/D layout -> lane holds column (l16), rows half*8 + r.
#pragma unroll
    for (int ti = 0; ti < 2; ++ti) {
        const int r0 = m0 + ti * 16 + half * 8;
#pragma unroll
        for (int tj = 0; tj < 2; ++tj) {
            const int col = n0 + tj * 16 + l16;
            const float bv = bias[col];
#pragma unroll
            for (int r = 0; r < 8; ++r)
                C[(size_t)(r0 + r) * (size_t)N + col] = acc[ti][tj][r] + bv;
        }
    }
}

// ---------------------------------------------------------------------------
// Scan layer 0: h[t] = tanh(pre[t] + h[t-1]*w); emit h as split bf16 for GEMM1.
// One thread per (b,h) column; software-pipelined so the (h-independent) load
// of step t+1 overlaps the tanh dependency chain of step t.
// ---------------------------------------------------------------------------
__global__ __launch_bounds__(256)
void scan_to_bf16(const float* __restrict__ pre,
                  const float* __restrict__ w_hh,
                  unsigned short* __restrict__ hh,
                  unsigned short* __restrict__ hl,
                  int T, int BH, int split) {
    const int col = blockIdx.x * blockDim.x + threadIdx.x;
    if (col >= BH) return;
    const float w = w_hh[col & (kH - 1)];
    float h = 0.f;
    size_t idx = (size_t)col;
    float p = pre[idx];
    for (int t = 0; t < T; ++t, idx += (size_t)BH) {
        const float pn = (t + 1 < T) ? pre[idx + (size_t)BH] : 0.f;  // prefetch
        h = tanhf(fmaf(h, w, p));
        const unsigned short hi = f32_to_bf16_rne(h);
        hh[idx] = hi;
        if (split) hl[idx] = f32_to_bf16_rne(h - bf16_to_f32(hi));
        p = pn;
    }
}

// ---------------------------------------------------------------------------
// Scan layer 1: in-place on d_out (pre -> out), plus zero the hidden tail.
// ---------------------------------------------------------------------------
__global__ __launch_bounds__(256)
void scan_final(float* __restrict__ buf,         // pre in, out1 out (in place)
                const float* __restrict__ w_hh,
                float* __restrict__ hidden,      // [2, BH] zeros
                int T, int BH) {
    const int col = blockIdx.x * blockDim.x + threadIdx.x;
    if (col >= BH) return;
    const float w = w_hh[col & (kH - 1)];
    float h = 0.f;
    size_t idx = (size_t)col;
    float p = buf[idx];
    for (int t = 0; t < T; ++t, idx += (size_t)BH) {
        const float pn = (t + 1 < T) ? buf[idx + (size_t)BH] : 0.f;  // prefetch
        h = tanhf(fmaf(h, w, p));
        buf[idx] = h;
        p = pn;
    }
    hidden[col] = 0.f;
    hidden[col + BH] = 0.f;
}

// ---------------------------------------------------------------------------
// Launch
// ---------------------------------------------------------------------------
extern "C" void kernel_launch(void* const* d_in, const int* in_sizes, int n_in,
                              void* d_out, int out_size, void* d_ws, size_t ws_size,
                              hipStream_t stream) {
    const float* x     = (const float*)d_in[0];   // [T,B,D]
    const float* W0    = (const float*)d_in[1];   // [H,D]
    const float* w_hh0 = (const float*)d_in[2];   // [H]
    const float* b0    = (const float*)d_in[3];   // [H]
    const float* W1    = (const float*)d_in[4];   // [H,H]
    const float* w_hh1 = (const float*)d_in[5];   // [H]
    const float* b1    = (const float*)d_in[6];   // [H]
    float* out = (float*)d_out;                   // [T,B,H] out1  +  [2,B,H] hidden

    const size_t szX  = (size_t)kM * kD * sizeof(unsigned short); // 64 MiB
    const size_t szW0 = (size_t)kH * kD * sizeof(unsigned short); // 1 MiB
    const size_t szW1 = (size_t)kH * kH * sizeof(unsigned short); // 2 MiB
    const size_t szHH = (size_t)kM * kH * sizeof(unsigned short); // 128 MiB

    const size_t need_split = 2 * (szX + szW0 + szW1 + szHH);     // ~390 MiB
    const bool   split      = ws_size >= need_split;

    char* w = (char*)d_ws;
    unsigned short *xh, *xl, *w0h, *w0l, *w1h, *w1l, *hh, *hl;
    size_t off = 0;
    if (split) {
        xh  = (unsigned short*)(w + off); off += szX;
        xl  = (unsigned short*)(w + off); off += szX;
        w0h = (unsigned short*)(w + off); off += szW0;
        w0l = (unsigned short*)(w + off); off += szW0;
        w1h = (unsigned short*)(w + off); off += szW1;
        w1l = (unsigned short*)(w + off); off += szW1;
        hh  = (unsigned short*)(w + off); off += szHH;
        hl  = (unsigned short*)(w + off); off += szHH;
    } else {
        xh  = (unsigned short*)(w + off); off += szX;
        w0h = (unsigned short*)(w + off); off += szW0;
        w1h = (unsigned short*)(w + off); off += szW1;
        hh  = (unsigned short*)(w + off); off += szHH;
        xl = xh; w0l = w0h; w1l = w1h; hl = hh;   // unused in non-split path
    }

    const int sflag = split ? 1 : 0;
    auto nblk = [](long long n) {
        long long b = (n + 255) / 256;
        return (int)(b > 16384 ? 16384 : b);
    };

    // 1) precision conversion
    const int nX  = (int)(kM * kD);
    const int nW0 = kH * kD;
    const int nW1 = kH * kH;
    to_bf16_split<<<nblk(nX),  256, 0, stream>>>(x,  xh,  xl,  nX,  sflag);
    to_bf16_split<<<nblk(nW0), 256, 0, stream>>>(W0, w0h, w0l, nW0, sflag);
    to_bf16_split<<<nblk(nW1), 256, 0, stream>>>(W1, w1h, w1l, nW1, sflag);

    // 2) GEMM0: pre0 = x @ W0^T + b0  -> staged in d_out (f32 [M,H])
    dim3 g0(kH / 128, (unsigned)(kM / 64));
    if (split)
        gemm_wmma_bf16<true ><<<g0, 256, 0, stream>>>(xh, xl, w0h, w0l, b0, out,
                                                      (int)kM, kH, kD);
    else
        gemm_wmma_bf16<false><<<g0, 256, 0, stream>>>(xh, xl, w0h, w0l, b0, out,
                                                      (int)kM, kH, kD);

    // 3) scan layer 0 -> h0 as split bf16 (input of GEMM1)
    scan_to_bf16<<<(int)(kBH / 256), 256, 0, stream>>>(out, w_hh0, hh, hl,
                                                       kT, (int)kBH, sflag);

    // 4) GEMM1: pre1 = h0 @ W1^T + b1 -> back into d_out
    dim3 g1(kH / 128, (unsigned)(kM / 64));
    if (split)
        gemm_wmma_bf16<true ><<<g1, 256, 0, stream>>>(hh, hl, w1h, w1l, b1, out,
                                                      (int)kM, kH, kH);
    else
        gemm_wmma_bf16<false><<<g1, 256, 0, stream>>>(hh, hl, w1h, w1l, b1, out,
                                                      (int)kM, kH, kH);

    // 5) scan layer 1 in place + zero hidden tail
    float* hidden = out + (size_t)kM * kH;
    scan_final<<<(int)(kBH / 256), 256, 0, stream>>>(out, w_hh1, hidden,
                                                     kT, (int)kBH);
}
